// CrossAttentionLayer_8675833938117
// MI455X (gfx1250) — matrix operand order repaired and checked by reading, not compiled, not run on hardware
//
#include <hip/hip_runtime.h>
#include <hip/hip_bf16.h>
#include <stdint.h>

// ---------------------------------------------------------------------------
// CDNA5 (gfx1250) cross-attention layer, bf16 WMMA pipeline.
// TDM (tensor_load_to_lds) staging with pad-on-load, double-buffered LDS,
// b128-vectorized fragment loads.
// ---------------------------------------------------------------------------

typedef __bf16 bf16;
typedef __attribute__((ext_vector_type(16))) __bf16 v16bf;
typedef __attribute__((ext_vector_type(8)))  float  v8f;
typedef __attribute__((ext_vector_type(8)))  __bf16 bf16x8;
typedef __attribute__((ext_vector_type(4)))  __bf16 bf16x4;

#define E_DIM   512
#define H_NUM   8
#define D_DIM   64
#define SEQ     2048
#define BATCH   4
#define M_TOTAL (BATCH * SEQ)        // 8192

#if __has_builtin(__builtin_amdgcn_tensor_load_to_lds)
#define HAVE_TDM 1
typedef __attribute__((ext_vector_type(4))) unsigned int u32x4;
typedef __attribute__((ext_vector_type(8))) int          i32x8;
typedef __attribute__((ext_vector_type(4))) int          i32x4;

// Tensor DMA descriptor per ISA 08 §8: group0 (count/lds_addr/global_addr/type),
// group1 (data_size=2B, pad_enable, pad_interval/amount, dims, strides).
// pad_iv: LDS pad inserted every 2^(pad_iv+3) bytes; pad_amt: (pad_amt+1)*4 bytes.
// This toolchain exposes the 6-arg builtin: (g0, g1, g2, g3, g_ext, cpol).
__device__ __forceinline__ void tdm_load(
    unsigned lds_addr, const void* gptr,
    unsigned tile_d0, unsigned tile_d1,
    unsigned tensor_d0, unsigned tensor_d1, unsigned stride0,
    unsigned pad_iv, unsigned pad_amt)
{
    const uint64_t ga = (uint64_t)(uintptr_t)gptr;
    u32x4 g0;
    g0[0] = 1u;                                    // count=1 (valid)
    g0[1] = lds_addr;                              // LDS byte offset
    g0[2] = (unsigned)ga;                          // global_addr[31:0]
    g0[3] = (unsigned)(ga >> 32) | (2u << 30);     // global_addr[56:32] | type=2
    i32x8 g1;
    g1[0] = (int)((1u << 16)                       // data_size = 2 bytes
                | (1u << 20)                       // pad_enable
                | (pad_iv  << 22)
                | (pad_amt << 25));
    g1[1] = (int)((tensor_d0 & 0xFFFFu) << 16);                      // dim0[15:0]
    g1[2] = (int)((tensor_d0 >> 16) | ((tensor_d1 & 0xFFFFu) << 16));// dim0 hi, dim1 lo
    g1[3] = (int)((tensor_d1 >> 16) | (tile_d0 << 16));              // dim1 hi, tile0
    g1[4] = (int)tile_d1;                                            // tile1 (tile2=0)
    g1[5] = (int)stride0;                                            // dim0_stride lo32
    g1[6] = 0;                                                       // stride hi, dim1_stride
    g1[7] = 0;
    const i32x4 z4 = {0, 0, 0, 0};
    const i32x8 z8 = {0, 0, 0, 0, 0, 0, 0, 0};
    __builtin_amdgcn_tensor_load_to_lds(g0, g1, z4, z4, z8, 0);
}
#else
#define HAVE_TDM 0
#endif

// WMMA fragment layouts as used here (contiguous k order, same for A and B), wave32:
//   A (16x32 bf16): lane L (M=L%16), elem e -> K = (L/16)*16 + e
//   B (32x16 bf16): lane L (N=L%16), elem e -> K = (L/16)*16 + e
//   C (16x16 f32):  lane L, vgpr r  -> M = (L/16)*8 + r, N = L%16.

__device__ __forceinline__ v16bf ld_frag_a(const bf16* p) {  // contiguous 16
    bf16x8 lo = *(const bf16x8*)p;
    bf16x8 hi = *(const bf16x8*)(p + 8);
    return __builtin_shufflevector(lo, hi, 0,1,2,3,4,5,6,7,8,9,10,11,12,13,14,15);
}
__device__ __forceinline__ v16bf ld_frag_b(const bf16* p) {  // contiguous 16
    bf16x8 lo = *(const bf16x8*)p;
    bf16x8 hi = *(const bf16x8*)(p + 8);
    return __builtin_shufflevector(lo, hi, 0,1,2,3,4,5,6,7,8,9,10,11,12,13,14,15);
}

// ---------------------------------------------------------------------------
// Kernel 1: LayerNorm (f32 -> bf16). One wave32 per 512-elem row.
// ---------------------------------------------------------------------------
__global__ __launch_bounds__(256)
void layernorm_kernel(const float* __restrict__ q, const float* __restrict__ kv,
                      const float* __restrict__ gq, const float* __restrict__ bq,
                      const float* __restrict__ gkv, const float* __restrict__ bkv,
                      bf16* __restrict__ qn, bf16* __restrict__ kvn)
{
    const int wid  = threadIdx.x >> 5;
    const int lane = threadIdx.x & 31;
    const int row  = blockIdx.x * 8 + wid;

    const float* src; const float* g; const float* bb; bf16* dst;
    if (row < M_TOTAL) {
        src = q  + (size_t)row * E_DIM;            g = gq;  bb = bq;
        dst = qn + (size_t)row * E_DIM;
    } else {
        const int r2 = row - M_TOTAL;
        src = kv  + (size_t)r2 * E_DIM;            g = gkv; bb = bkv;
        dst = kvn + (size_t)r2 * E_DIM;
    }

    float4 x[4];
    float s = 0.f, ss = 0.f;
#pragma unroll
    for (int j = 0; j < 4; ++j) {
        x[j] = *(const float4*)&src[j * 128 + lane * 4];
        s  += x[j].x + x[j].y + x[j].z + x[j].w;
        ss += x[j].x * x[j].x + x[j].y * x[j].y + x[j].z * x[j].z + x[j].w * x[j].w;
    }
#pragma unroll
    for (int off = 16; off > 0; off >>= 1) {
        s  += __shfl_xor(s,  off, 32);
        ss += __shfl_xor(ss, off, 32);
    }
    const float mu  = s * (1.0f / E_DIM);
    const float var = ss * (1.0f / E_DIM) - mu * mu;
    const float inv = rsqrtf(var + 1e-5f);

#pragma unroll
    for (int j = 0; j < 4; ++j) {
        const int col = j * 128 + lane * 4;
        bf16x4 y;
        y[0] = (bf16)((x[j].x - mu) * inv * g[col + 0] + bb[col + 0]);
        y[1] = (bf16)((x[j].y - mu) * inv * g[col + 1] + bb[col + 1]);
        y[2] = (bf16)((x[j].z - mu) * inv * g[col + 2] + bb[col + 2]);
        y[3] = (bf16)((x[j].w - mu) * inv * g[col + 3] + bb[col + 3]);
        *(bf16x4*)&dst[col] = y;
    }
}

// ---------------------------------------------------------------------------
// Kernel 2: weights f32 -> bf16 K-panel fragment-major layout:
//   Wp[((k>>5)*E + n)*32 + (k&31)]
// ---------------------------------------------------------------------------
__global__ __launch_bounds__(256)
void weight_panel_kernel(const float* __restrict__ W, bf16* __restrict__ Wp)
{
    __shared__ __align__(16) bf16 T[128][40];
    const int k0  = blockIdx.x * 32;
    const int n0  = blockIdx.y * 128;
    const int tid = threadIdx.x;

#pragma unroll
    for (int i = 0; i < 4; ++i) {
        const int v = (tid + i * 256) * 4;
        const int k = v >> 7, n = v & 127;
        float4 x = *(const float4*)&W[(size_t)(k0 + k) * E_DIM + n0 + n];
        T[n + 0][k] = (bf16)x.x;
        T[n + 1][k] = (bf16)x.y;
        T[n + 2][k] = (bf16)x.z;
        T[n + 3][k] = (bf16)x.w;
    }
    __syncthreads();

    const int n = tid >> 1, half = tid & 1;
    bf16x8 lo = *(const bf16x8*)&T[n][half * 16];
    bf16x8 hi = *(const bf16x8*)&T[n][half * 16 + 8];
    bf16* dst = &Wp[(((size_t)(k0 >> 5)) * E_DIM + n0 + n) * 32 + half * 16];
    *(bf16x8*)dst       = lo;
    *(bf16x8*)(dst + 8) = hi;
}

// ---------------------------------------------------------------------------
// Kernel 3: tiled bf16 GEMM  out = A(MxE) @ W(ExE) + bias, W in panel layout.
// Block tile 32x128, 8 waves (2Mx4N), wave = 16x32 C tile, K step 32.
// Tiles staged by TDM (wave 0 issues, TENSORcnt-synced), double-buffered.
// mode 0: f32 row-major. mode 1: bf16 [B,H,seq,D]. mode 2: bf16 [B,H,D,seq].
// ---------------------------------------------------------------------------
__global__ __launch_bounds__(256)
void gemm_bf16_kernel(const bf16* __restrict__ A, const bf16* __restrict__ Wp,
                      const float* __restrict__ bias,
                      bf16* __restrict__ outHead, float* __restrict__ outF32,
                      int mode)
{
    __shared__ __align__(16) bf16 As[2][32][40];    // 64B rows + 16B TDM pad
    __shared__ __align__(16) bf16 BsF[2][128][40];  // 64B rows + 16B TDM pad

    const int tid   = threadIdx.x;
    const int lane  = tid & 31;
    const int wid   = tid >> 5;
    const int waveM = wid & 1;
    const int waveN = wid >> 1;
    const int mBase = blockIdx.x * 32;
    const int nBase = blockIdx.y * 128;
    const int lm = lane & 15, lh = lane >> 4;

    auto stage = [&](int buf, int k0) {
#if HAVE_TDM
        if (tid < 32) {   // one wave issues the DMA (TDM is per-wave)
            tdm_load((unsigned)(uintptr_t)&As[buf][0][0],
                     &A[(size_t)mBase * E_DIM + k0],
                     /*tile*/ 32, 32, /*tensor*/ E_DIM, M_TOTAL,
                     /*stride0*/ E_DIM, /*pad 64B->16B*/ 3, 3);
            tdm_load((unsigned)(uintptr_t)&BsF[buf][0][0],
                     Wp + ((size_t)(k0 >> 5) * E_DIM + nBase) * 32,
                     /*tile*/ 4096, 1, /*tensor*/ 4096, 1,
                     /*stride0*/ 4096, /*pad 64B->16B*/ 3, 3);
        }
#else
        {   // manual fallback: A vec4/thread, B contiguous b128 copies
            const int v = tid * 4;
            const int row = v >> 5, col = v & 31;
            *(bf16x4*)&As[buf][row][col] =
                *(const bf16x4*)&A[(size_t)(mBase + row) * E_DIM + k0 + col];
            const bf16* wp = Wp + ((size_t)(k0 >> 5) * E_DIM + nBase) * 32;
            bf16x8 lo = *(const bf16x8*)&wp[tid * 16];
            bf16x8 hi = *(const bf16x8*)&wp[tid * 16 + 8];
            bf16* d = &BsF[buf][tid >> 1][(tid & 1) * 16];
            *(bf16x8*)d       = lo;
            *(bf16x8*)(d + 8) = hi;
        }
#endif
    };

    v8f c0 = {}, c1 = {};
    stage(0, 0);

    for (int k0 = 0; k0 < E_DIM; k0 += 32) {
        const int cur = (k0 >> 5) & 1;
#if HAVE_TDM
        if (tid < 32) __builtin_amdgcn_s_wait_tensorcnt(0);
#endif
        __syncthreads();
        if (k0 + 32 < E_DIM) stage(cur ^ 1, k0 + 32);

        const v16bf a  = ld_frag_a(&As[cur][waveM * 16 + lm][lh * 16]);
        const int   n0 = waveN * 32 + lm;
        const v16bf b0 = ld_frag_b(&BsF[cur][n0][lh * 16]);
        const v16bf b1 = ld_frag_b(&BsF[cur][n0 + 16][lh * 16]);
        c0 = __builtin_amdgcn_wmma_f32_16x16x32_bf16(false, a, false, b0, (short)0, c0, false, false);
        c1 = __builtin_amdgcn_wmma_f32_16x16x32_bf16(false, a, false, b1, (short)0, c1, false, false);
    }

#pragma unroll
    for (int t = 0; t < 2; ++t) {
        const v8f& c = t ? c1 : c0;
#pragma unroll
        for (int r = 0; r < 8; ++r) {
            const int m = mBase + waveM * 16 + lh * 8 + r;
            const int n = nBase + waveN * 32 + t * 16 + lm;
            const float val = c[r] + bias[n];
            const int b = m >> 11, qq = m & (SEQ - 1);
            const int h = n >> 6,  d  = n & (D_DIM - 1);
            if (mode == 0) {
                outF32[(size_t)m * E_DIM + n] = val;
            } else if (mode == 1) {     // [B,H,seq,D]
                outHead[(((size_t)b * H_NUM + h) * SEQ + qq) * D_DIM + d] = (bf16)val;
            } else {                    // [B,H,D,seq]  (V transposed)
                outHead[(((size_t)b * H_NUM + h) * D_DIM + d) * SEQ + qq] = (bf16)val;
            }
        }
    }
}

// ---------------------------------------------------------------------------
// Kernel 4: masked flash attention per (b,h). 4 waves/block, 16 q-rows/wave,
// 32 keys/iteration. K/V tiles staged by TDM (double-buffered); masks
// bit-packed via u64 ops (byte c of row M: bit0=mask(c), bit1=mask(c+16)).
// ---------------------------------------------------------------------------
__global__ __launch_bounds__(128)
void attention_kernel(const bf16* __restrict__ qh, const bf16* __restrict__ kh,
                      const bf16* __restrict__ vhT,
                      const unsigned char* __restrict__ kv_mask,
                      const unsigned char* __restrict__ sparse_mask,
                      bf16* __restrict__ oh)
{
    __shared__ __align__(16) bf16 Ks[2][32][72];     // 128B rows + 16B TDM pad
    __shared__ __align__(16) bf16 VsT[2][64][40];    //  64B rows + 16B TDM pad
    __shared__ __align__(16) bf16 Ps[4][16][40];     // per-wave P, [row][key]
    __shared__ __align__(16) unsigned char Ms2[2][4][16][16];

    const int tid = threadIdx.x, lane = tid & 31, w = tid >> 5;
    const int lm = lane & 15, lh = lane >> 4;
    const int b = blockIdx.z, h = blockIdx.y;
    const int qBase = blockIdx.x * 64 + w * 16;
    const float NEG_INF = -__builtin_huge_valf();
    const float SCALE = 0.125f;                      // 64^-0.5

    const size_t headOff  = ((size_t)b * H_NUM + h) * SEQ;    // rows in [B,H,seq,D]
    const size_t headOffD = ((size_t)b * H_NUM + h) * D_DIM;  // rows in [B,H,D,seq]

    auto stage_kv = [&](int buf, int kb) {
#if HAVE_TDM
        if (tid < 32) {
            // K tile: 32 keys x 64 d = 2048 contiguous elems; pad 16B per 128B.
            tdm_load((unsigned)(uintptr_t)&Ks[buf][0][0],
                     &kh[(headOff + kb) * D_DIM],
                     2048, 1, 2048, 1, 2048, 4, 3);
            // V^T tile: 64 rows (d) x 32 keys, row stride SEQ; pad 16B per 64B.
            tdm_load((unsigned)(uintptr_t)&VsT[buf][0][0],
                     &vhT[headOffD * SEQ + kb],
                     32, 64, SEQ, (unsigned)(BATCH * H_NUM * D_DIM), SEQ, 3, 3);
        }
#else
        {
            const int key = tid >> 2, q4 = tid & 3;
            const bf16* src = &kh[(headOff + kb + key) * D_DIM + q4 * 16];
            bf16x8 lo = *(const bf16x8*)src;
            bf16x8 hi = *(const bf16x8*)(src + 8);
            bf16* d = &Ks[buf][key][q4 * 16];
            *(bf16x8*)d       = lo;
            *(bf16x8*)(d + 8) = hi;
        }
        {
            const int dd = tid >> 1, half = tid & 1;
            const bf16* src = &vhT[(headOffD + dd) * SEQ + kb + half * 16];
            bf16x8 lo = *(const bf16x8*)src;
            bf16x8 hi = *(const bf16x8*)(src + 8);
            bf16* d = &VsT[buf][dd][half * 16];
            *(bf16x8*)d       = lo;
            *(bf16x8*)(d + 8) = hi;
        }
#endif
    };
    auto stage_mask = [&](int buf, int kb) {
        if (lane < 16) {
            const uint64_t* sm = (const uint64_t*)(sparse_mask +
                ((size_t)b * SEQ + qBase + lane) * SEQ + kb);
            const uint64_t* km = (const uint64_t*)(kv_mask + (size_t)b * SEQ + kb);
            const uint64_t M1 = 0x0101010101010101ull;
            const uint64_t r0 = (sm[0] & km[0] & M1) | ((sm[2] & km[2] & M1) << 1);
            const uint64_t r1 = (sm[1] & km[1] & M1) | ((sm[3] & km[3] & M1) << 1);
            *(uint64_t*)&Ms2[buf][w][lane][0] = r0;
            *(uint64_t*)&Ms2[buf][w][lane][8] = r1;
        }
    };

    // Q fragments for the two D-chunks, loaded once as b128 pairs.
    const bf16* qp = qh + (headOff + qBase + lm) * D_DIM;
    const v16bf qa0 = ld_frag_a(qp + lh * 16);
    const v16bf qa1 = ld_frag_a(qp + 32 + lh * 16);

    float mrow[8], lrow[8];
    v8f acc[4] = {};
#pragma unroll
    for (int r = 0; r < 8; ++r) { mrow[r] = NEG_INF; lrow[r] = 0.f; }

    stage_kv(0, 0);
    stage_mask(0, 0);

    for (int kb = 0; kb < SEQ; kb += 32) {
        const int cur = (kb >> 5) & 1;
#if HAVE_TDM
        if (tid < 32) __builtin_amdgcn_s_wait_tensorcnt(0);
#endif
        __syncthreads();
        if (kb + 32 < SEQ) { stage_kv(cur ^ 1, kb + 32); stage_mask(cur ^ 1, kb + 32); }
        if (w == 1 && kb + 64 < SEQ)   // keep L2 warm two tiles ahead
            __builtin_prefetch(&kh[(headOff + kb + 64 + lane) * D_DIM], 0, 0);

        // S = Q @ K^T : two 16-key tiles x two D-chunks.
        v8f s0 = {}, s1 = {};
        {
            const v16bf bt0c0 = ld_frag_b(&Ks[cur][lm][lh * 16]);
            const v16bf bt1c0 = ld_frag_b(&Ks[cur][16 + lm][lh * 16]);
            s0 = __builtin_amdgcn_wmma_f32_16x16x32_bf16(false, qa0, false, bt0c0, (short)0, s0, false, false);
            s1 = __builtin_amdgcn_wmma_f32_16x16x32_bf16(false, qa0, false, bt1c0, (short)0, s1, false, false);
            const v16bf bt0c1 = ld_frag_b(&Ks[cur][lm][32 + lh * 16]);
            const v16bf bt1c1 = ld_frag_b(&Ks[cur][16 + lm][32 + lh * 16]);
            s0 = __builtin_amdgcn_wmma_f32_16x16x32_bf16(false, qa1, false, bt0c1, (short)0, s0, false, false);
            s1 = __builtin_amdgcn_wmma_f32_16x16x32_bf16(false, qa1, false, bt1c1, (short)0, s1, false, false);
        }

        // Online softmax. Row M = lh*8+r lives across 16 lanes sharing lh.
#pragma unroll
        for (int r = 0; r < 8; ++r) {
            const int M = lh * 8 + r;
            const unsigned char mb = Ms2[cur][w][M][lm];
            const float v0 = (mb & 1) ? s0[r] * SCALE : NEG_INF;
            const float v1 = (mb & 2) ? s1[r] * SCALE : NEG_INF;
            float tm = fmaxf(v0, v1);
#pragma unroll
            for (int off = 8; off > 0; off >>= 1) tm = fmaxf(tm, __shfl_xor(tm, off, 32));
            const float nm = fmaxf(mrow[r], tm);
            const float alpha = (nm == mrow[r]) ? 1.f : __expf(mrow[r] - nm);
            const float p0 = (mb & 1) ? __expf(v0 - nm) : 0.f;
            const float p1 = (mb & 2) ? __expf(v1 - nm) : 0.f;
            float ps = p0 + p1;
#pragma unroll
            for (int off = 8; off > 0; off >>= 1) ps += __shfl_xor(ps, off, 32);
            lrow[r] = lrow[r] * alpha + ps;
            mrow[r] = nm;
#pragma unroll
            for (int dt = 0; dt < 4; ++dt) acc[dt][r] *= alpha;
            Ps[w][M][lm]      = (bf16)p0;     // C layout -> LDS for transpose
            Ps[w][M][16 + lm] = (bf16)p1;
        }
        __builtin_amdgcn_wave_barrier();      // fence LDS store->load within wave

        // O += P @ V
        {
            const v16bf ap = ld_frag_a(&Ps[w][lm][lh * 16]);
#pragma unroll
            for (int dt = 0; dt < 4; ++dt) {
                const v16bf bv = ld_frag_b(&VsT[cur][dt * 16 + lm][lh * 16]);
                acc[dt] = __builtin_amdgcn_wmma_f32_16x16x32_bf16(false, ap, false, bv,
                                                                  (short)0, acc[dt], false, false);
            }
        }
    }

    // Normalize (rows with no valid key -> 0) and store interleaved [B,Q,E].
#pragma unroll
    for (int r = 0; r < 8; ++r) {
        const int qrow = qBase + lh * 8 + r;
        const float linv = (lrow[r] > 0.f) ? 1.0f / lrow[r] : 0.f;
#pragma unroll
        for (int dt = 0; dt < 4; ++dt) {
            const int d = dt * 16 + lm;
            oh[((size_t)b * SEQ + qrow) * E_DIM + h * D_DIM + d] = (bf16)(acc[dt][r] * linv);
        }
    }
}

// ---------------------------------------------------------------------------
// Host launcher
// ---------------------------------------------------------------------------
extern "C" void kernel_launch(void* const* d_in, const int* in_sizes, int n_in,
                              void* d_out, int out_size, void* d_ws, size_t ws_size,
                              hipStream_t stream)
{
    (void)in_sizes; (void)n_in; (void)out_size; (void)ws_size;

    const float*         query       = (const float*)d_in[0];
    const float*         key_value   = (const float*)d_in[1];
    const unsigned char* kv_mask     = (const unsigned char*)d_in[2];  // bool (1B)
    const unsigned char* sparse_mask = (const unsigned char*)d_in[3];  // bool (1B)
    const float* ln_q_g  = (const float*)d_in[4];
    const float* ln_q_b  = (const float*)d_in[5];
    const float* ln_kv_g = (const float*)d_in[6];
    const float* ln_kv_b = (const float*)d_in[7];
    const float* Wq = (const float*)d_in[8];   const float* bq = (const float*)d_in[9];
    const float* Wk = (const float*)d_in[10];  const float* bk = (const float*)d_in[11];
    const float* Wv = (const float*)d_in[12];  const float* bv = (const float*)d_in[13];
    const float* Wo = (const float*)d_in[14];  const float* bo = (const float*)d_in[15];

    char* ws = (char*)d_ws;
    size_t off = 0;
    auto carve = [&](size_t bytes) -> void* {
        void* p = ws + off;
        off += (bytes + 255) & ~(size_t)255;
        return p;
    };
    const size_t SZ_ACT = (size_t)M_TOTAL * E_DIM * sizeof(bf16);  // 8 MB
    const size_t SZ_W   = (size_t)E_DIM * E_DIM * sizeof(bf16);    // 0.5 MB

    bf16* qn  = (bf16*)carve(SZ_ACT);
    bf16* kvn = (bf16*)carve(SZ_ACT);
    bf16* wqp = (bf16*)carve(SZ_W);
    bf16* wkp = (bf16*)carve(SZ_W);
    bf16* wvp = (bf16*)carve(SZ_W);
    bf16* wop = (bf16*)carve(SZ_W);
    bf16* qhp = (bf16*)carve(SZ_ACT);   // [B,H,Q,D]
    bf16* khp = (bf16*)carve(SZ_ACT);   // [B,H,K,D]
    bf16* vhT = (bf16*)carve(SZ_ACT);   // [B,H,D,K]
    bf16* ohp = qn;                     // qn dead after Q projection: reuse

    layernorm_kernel<<<dim3((2 * M_TOTAL) / 8), 256, 0, stream>>>(
        query, key_value, ln_q_g, ln_q_b, ln_kv_g, ln_kv_b, qn, kvn);

    const dim3 wGrid(E_DIM / 32, E_DIM / 128);
    weight_panel_kernel<<<wGrid, 256, 0, stream>>>(Wq, wqp);
    weight_panel_kernel<<<wGrid, 256, 0, stream>>>(Wk, wkp);
    weight_panel_kernel<<<wGrid, 256, 0, stream>>>(Wv, wvp);
    weight_panel_kernel<<<wGrid, 256, 0, stream>>>(Wo, wop);

    const dim3 gGrid(M_TOTAL / 32, E_DIM / 128);
    gemm_bf16_kernel<<<gGrid, 256, 0, stream>>>(qn,  wqp, bq, qhp, nullptr, 1);
    gemm_bf16_kernel<<<gGrid, 256, 0, stream>>>(kvn, wkp, bk, khp, nullptr, 1);
    gemm_bf16_kernel<<<gGrid, 256, 0, stream>>>(kvn, wvp, bv, vhT, nullptr, 2);

    attention_kernel<<<dim3(SEQ / 64, H_NUM, BATCH), 128, 0, stream>>>(
        qhp, khp, vhT, kv_mask, sparse_mask, ohp);

    gemm_bf16_kernel<<<gGrid, 256, 0, stream>>>(ohp, wop, bo, nullptr, (float*)d_out, 0);
}
